// Qwen3VisionTransformer_74259984548247
// MI455X (gfx1250) — matrix operand order repaired and checked by reading, not compiled, not run on hardware
//
#include <hip/hip_runtime.h>
#include <stdint.h>

#define S_TOK 16384
#define HID   1280
#define NHEAD 16
#define HD    80
#define NLAYER 8
#define FFD   3456
#define WINSZ 64
#define PVD   1536
#define MROWS 4096
#define MHID  5120
#define OUTD  2048

typedef __attribute__((ext_vector_type(16))) __bf16 v16bf;
typedef __attribute__((ext_vector_type(8)))  float  v8f;

struct __align__(16) U4 { unsigned a[4]; };
union FragU { v16bf v; U4 q[2]; };
union RowU  { U4 q[2]; unsigned short s[16]; };

__device__ __forceinline__ unsigned short f2bf(float f) {
  union { float f; unsigned u; } x; x.f = f;
  unsigned r = x.u + 0x7FFFu + ((x.u >> 16) & 1u);
  return (unsigned short)(r >> 16);
}
__device__ __forceinline__ float bf2f(unsigned short h) {
  union { unsigned u; float f; } x; x.u = ((unsigned)h) << 16; return x.f;
}
__device__ __forceinline__ float gelu_tanh(float x) {
  float x3 = x * x * x;
  float t = tanhf(0.7978845608028654f * (x + 0.044715f * x3));
  return 0.5f * x * (1.0f + t);
}
__device__ __forceinline__ v8f wmma_bf16(v16bf a, v16bf b, v8f c) {
  return __builtin_amdgcn_wmma_f32_16x16x32_bf16(false, a, false, b, (short)0, c, false, false);
}
// A-fragment (16x32 bf16): lane<16 -> row M=lane, K (kk+0..7, kk+16..23); lane>=16 -> K(+8,+24)
__device__ __forceinline__ v16bf load_a_frag(const unsigned short* p, int stride, int row0, int kk) {
  int lane = threadIdx.x & 31;
  int hf = lane >> 4;
  const unsigned short* base = p + (row0 + (lane & 15)) * stride + kk + hf * 8;
  FragU u;
  u.q[0] = *(const U4*)(base);
  u.q[1] = *(const U4*)(base + 16);
  return u.v;
}
// B-fragment (32x16 bf16) from K-major (transposed) LDS: lane col N, K (kk+0..7, kk+16..23); lane>=16 -> K(+8,+24)
__device__ __forceinline__ v16bf load_b_frag(const unsigned short* p, int stride, int col0, int kk) {
  int lane = threadIdx.x & 31;
  int hf = lane >> 4;
  const unsigned short* base = p + (col0 + (lane & 15)) * stride + kk + hf * 8;
  FragU u;
  u.q[0] = *(const U4*)(base);
  u.q[1] = *(const U4*)(base + 16);
  return u.v;
}

// ---------------- elementwise f32 -> bf16 ----------------
__global__ __launch_bounds__(256) void cvt_kernel(const float* __restrict__ in,
                                                  unsigned short* __restrict__ out, size_t n) {
  size_t i = (size_t)blockIdx.x * 256 + threadIdx.x;
  if (i < n) out[i] = f2bf(in[i]);
}

// ---------------- layernorm row (N=1280) -> bf16 ----------------
__global__ __launch_bounds__(256) void ln_bf16_kernel(const float* __restrict__ x,
    const float* __restrict__ w, const float* __restrict__ b,
    unsigned short* __restrict__ out, int N) {
  __shared__ float red[16];
  int row = blockIdx.x;
  const float* xr = x + (size_t)row * N;
  float s = 0.f, s2 = 0.f;
  for (int i = threadIdx.x; i < N; i += 256) { float v = xr[i]; s += v; s2 += v * v; }
  for (int o = 16; o > 0; o >>= 1) { s += __shfl_down(s, o, 32); s2 += __shfl_down(s2, o, 32); }
  int wv = threadIdx.x >> 5, ln = threadIdx.x & 31;
  if (ln == 0) { red[wv] = s; red[8 + wv] = s2; }
  __syncthreads();
  if (threadIdx.x == 0) {
    float a = 0.f, c = 0.f;
    for (int i = 0; i < 8; i++) { a += red[i]; c += red[8 + i]; }
    red[0] = a; red[8] = c;
  }
  __syncthreads();
  float mean = red[0] / (float)N;
  float var  = red[8] / (float)N - mean * mean;
  float rs = rsqrtf(var + 1e-6f);
  for (int i = threadIdx.x; i < N; i += 256) {
    float v = (xr[i] - mean) * rs * w[i] + b[i];
    out[(size_t)row * N + i] = f2bf(v);
  }
}

// ---------------- in-place RoPE on bf16 qkv (q and k halves) ----------------
__global__ __launch_bounds__(256) void rope_kernel(unsigned short* __restrict__ qkv,
    const float* __restrict__ cosT, const float* __restrict__ sinT) {
  int idx = blockIdx.x * 256 + threadIdx.x;       // S * 2 * 16 * 40 threads
  int d = idx % 40;
  int t = idx / 40;
  int hh = t % NHEAD; t /= NHEAD;
  int which = t & 1;  int s = t >> 1;
  size_t base = (size_t)s * (3 * HID) + (size_t)which * HID + hh * HD;
  float x1 = bf2f(qkv[base + d]);
  float x2 = bf2f(qkv[base + d + 40]);
  float c = cosT[(size_t)s * HD + d];
  float sn = sinT[(size_t)s * HD + d];
  qkv[base + d]      = f2bf(x1 * c - x2 * sn);
  qkv[base + d + 40] = f2bf(x2 * c + x1 * sn);
}

// ---------------- bf16 GEMM: C(MxN) = A(MxK) * B(KxN) + bias, fused epilogues ----------------
// EPI 0: f32 out = acc+bias ; 1: f32 out = acc+bias+res ; 2: bf16 out ; 3: bf16 gelu(out)
template<int EPI>
__global__ __launch_bounds__(256) void gemm_bf16_kernel(
    const unsigned short* __restrict__ A, const unsigned short* __restrict__ B,
    const float* __restrict__ bias, const float* __restrict__ res,
    float* __restrict__ outF, unsigned short* __restrict__ outB,
    int M, int N, int K) {
  __shared__ unsigned short As[128 * 72];   // row-major, stride 72
  __shared__ unsigned short Bs[128 * 72];   // N-major (transposed), stride 72
  const int tid  = threadIdx.x;
  const int lane = tid & 31;
  const int wave = tid >> 5;
  const int wm = wave & 1, wn = wave >> 1;
  const int tileM = blockIdx.y * 128;
  const int tileN = blockIdx.x * 128;

  v8f acc[4][2];
  const v8f vzero = {0.f,0.f,0.f,0.f,0.f,0.f,0.f,0.f};
#pragma unroll
  for (int i = 0; i < 4; i++)
#pragma unroll
    for (int j = 0; j < 2; j++) acc[i][j] = vzero;

  const int aRow = tid >> 1;            // 0..127
  const int aSeg = (tid & 1) * 32;      // 0 or 32 (elements)
  const int bNg  = (tid & 7) * 16;      // n-group start
  const int bKp  = (tid >> 3) * 2;      // k pair 0..62

  // LDS byte address (flat LDS aperture: low 32 bits == LDS offset) for this
  // thread's 64-byte slice of the A tile (async DMA destination).
  const unsigned aLds = (unsigned)(unsigned long long)&As[aRow * 72 + aSeg];

  for (int k0 = 0; k0 < K; k0 += 64) {
    // ---- B tile: global -> regs (needs transpose, keep VGPR path)
    const unsigned short* bg0 = B + (size_t)(k0 + bKp) * N + tileN + bNg;
    const unsigned short* bg1 = bg0 + N;
    RowU r0, r1;
    r0.q[0] = *(const U4*)(bg0); r0.q[1] = *(const U4*)(bg0 + 8);
    r1.q[0] = *(const U4*)(bg1); r1.q[1] = *(const U4*)(bg1 + 8);
    if (k0 + 64 < K) {
      __builtin_prefetch(bg0 + (size_t)64 * N, 0, 1);
    }
    __syncthreads();   // previous compute done before LDS overwrite

    // ---- A tile: async DMA global -> LDS (64B per thread, ASYNCcnt path)
    {
      unsigned long long ga =
          (unsigned long long)(A + (size_t)(tileM + aRow) * K + k0 + aSeg);
      asm volatile(
        "global_load_async_to_lds_b128 %0, %1, off\n\t"
        "global_load_async_to_lds_b128 %0, %1, off offset:16\n\t"
        "global_load_async_to_lds_b128 %0, %1, off offset:32\n\t"
        "global_load_async_to_lds_b128 %0, %1, off offset:48"
        :: "v"(aLds), "v"(ga) : "memory");
    }

    // ---- B tile: regs -> LDS transposed
#pragma unroll
    for (int i = 0; i < 16; i++) {
      unsigned v = (unsigned)r0.s[i] | ((unsigned)r1.s[i] << 16);
      *(unsigned*)&Bs[(bNg + i) * 72 + bKp] = v;
    }

    asm volatile("s_wait_asynccnt 0" ::: "memory");   // this wave's DMA done
    __syncthreads();                                   // all waves staged

    // ---- compute
#pragma unroll
    for (int kk = 0; kk < 64; kk += 32) {
      v16bf af[4], bfr[2];
#pragma unroll
      for (int mt = 0; mt < 4; mt++) af[mt] = load_a_frag(As, 72, wm * 64 + mt * 16, kk);
#pragma unroll
      for (int nt = 0; nt < 2; nt++) bfr[nt] = load_b_frag(Bs, 72, wn * 32 + nt * 16, kk);
#pragma unroll
      for (int mt = 0; mt < 4; mt++)
#pragma unroll
        for (int nt = 0; nt < 2; nt++)
          acc[mt][nt] = wmma_bf16(af[mt], bfr[nt], acc[mt][nt]);
    }
  }

  // ---- epilogue
  const int hl = lane >> 4, nc = lane & 15;
#pragma unroll
  for (int mt = 0; mt < 4; mt++) {
#pragma unroll
    for (int nt = 0; nt < 2; nt++) {
      int col = tileN + wn * 32 + nt * 16 + nc;
      float bv = bias[col];
#pragma unroll
      for (int j = 0; j < 8; j++) {
        int row = tileM + wm * 64 + mt * 16 + hl * 8 + j;
        size_t off = (size_t)row * N + col;
        float v = acc[mt][nt][j] + bv;
        if (EPI == 0) outF[off] = v;
        else if (EPI == 1) outF[off] = v + res[off];
        else if (EPI == 2) outB[off] = f2bf(v);
        else outB[off] = f2bf(gelu_tanh(v));
      }
    }
  }
}

// ---------------- windowed attention: one WG per (window, head) ----------------
__global__ __launch_bounds__(128) void attn_kernel(const unsigned short* __restrict__ qkv,
                                                   unsigned short* __restrict__ outB) {
  __shared__ unsigned short Qs[64 * 104];  // [token][feat], K padded to 96
  __shared__ unsigned short Ks[64 * 104];
  __shared__ unsigned short Vt[80 * 72];   // [feat][token]
  __shared__ unsigned short Ps[64 * 72];   // softmax probs bf16
  __shared__ float Sf[64 * 68];            // scores f32
  const int win = blockIdx.x >> 4;
  const int hd  = blockIdx.x & 15;
  const int tid = threadIdx.x;
  const int lane = tid & 31;
  const int wave = tid >> 5;   // 0..3 -> 16-row M slab
  const size_t gbase = (size_t)win * WINSZ * (3 * HID) + (size_t)hd * HD;

  for (int i = tid; i < 64 * 40; i += 128) {
    int r = i / 40, d2 = (i % 40) * 2;
    size_t g = gbase + (size_t)r * (3 * HID) + d2;
    unsigned q2 = *(const unsigned*)(qkv + g);
    unsigned k2 = *(const unsigned*)(qkv + g + HID);
    unsigned v2 = *(const unsigned*)(qkv + g + 2 * HID);
    *(unsigned*)&Qs[r * 104 + d2] = q2;
    *(unsigned*)&Ks[r * 104 + d2] = k2;
    Vt[d2 * 72 + r]       = (unsigned short)(v2 & 0xFFFFu);
    Vt[(d2 + 1) * 72 + r] = (unsigned short)(v2 >> 16);
  }
  for (int i = tid; i < 64 * 8; i += 128) {   // zero K-pad feats 80..95
    int r = i / 8, d2 = 80 + (i % 8) * 2;
    *(unsigned*)&Qs[r * 104 + d2] = 0u;
    *(unsigned*)&Ks[r * 104 + d2] = 0u;
  }
  __syncthreads();

  const float scale = 0.11180339887498949f;   // 80^-0.5
  const v8f vzero = {0.f,0.f,0.f,0.f,0.f,0.f,0.f,0.f};
#pragma unroll
  for (int nt = 0; nt < 4; nt++) {
    v8f acc = vzero;
#pragma unroll
    for (int kk = 0; kk < 96; kk += 32) {
      v16bf a = load_a_frag(Qs, 104, wave * 16, kk);
      v16bf b = load_b_frag(Ks, 104, nt * 16, kk);
      acc = wmma_bf16(a, b, acc);
    }
    int hl = lane >> 4, nc = lane & 15;
#pragma unroll
    for (int j = 0; j < 8; j++)
      Sf[(wave * 16 + hl * 8 + j) * 68 + nt * 16 + nc] = acc[j] * scale;
  }
  __syncthreads();

  if (tid < 64) {
    float m = -1e30f;
    for (int c = 0; c < 64; c++) m = fmaxf(m, Sf[tid * 68 + c]);
    float sum = 0.f;
    for (int c = 0; c < 64; c++) sum += __expf(Sf[tid * 68 + c] - m);
    float inv = 1.0f / sum;
    for (int c = 0; c < 64; c++)
      Ps[tid * 72 + c] = f2bf(__expf(Sf[tid * 68 + c] - m) * inv);
  }
  __syncthreads();

#pragma unroll
  for (int nt = 0; nt < 5; nt++) {
    v8f acc = vzero;
#pragma unroll
    for (int kk = 0; kk < 64; kk += 32) {
      v16bf a = load_a_frag(Ps, 72, wave * 16, kk);
      v16bf b = load_b_frag(Vt, 72, nt * 16, kk);
      acc = wmma_bf16(a, b, acc);
    }
    int hl = lane >> 4, nc = lane & 15;
    int f = nt * 16 + nc;     // < 80 always
#pragma unroll
    for (int j = 0; j < 8; j++) {
      int row = wave * 16 + hl * 8 + j;
      size_t off = ((size_t)win * WINSZ + row) * HID + (size_t)hd * HD + f;
      outB[off] = f2bf(acc[j]);
    }
  }
}

extern "C" void kernel_launch(void* const* d_in, const int* in_sizes, int n_in,
                              void* d_out, int out_size, void* d_ws, size_t ws_size,
                              hipStream_t stream) {
  const float* x     = (const float*)d_in[0];
  const float* cosT  = (const float*)d_in[1];
  const float* sinT  = (const float*)d_in[2];
  const float* pe_w  = (const float*)d_in[4];
  const float* pe_b  = (const float*)d_in[5];
  const float* ln1_w = (const float*)d_in[6];
  const float* ln1_b = (const float*)d_in[7];
  const float* qkv_w = (const float*)d_in[8];
  const float* qkv_b = (const float*)d_in[9];
  const float* proj_w= (const float*)d_in[10];
  const float* proj_b= (const float*)d_in[11];
  const float* ln2_w = (const float*)d_in[12];
  const float* ln2_b = (const float*)d_in[13];
  const float* fc1_w = (const float*)d_in[14];
  const float* fc1_b = (const float*)d_in[15];
  const float* fc2_w = (const float*)d_in[16];
  const float* fc2_b = (const float*)d_in[17];
  const float* mlnw  = (const float*)d_in[18];
  const float* mlnb  = (const float*)d_in[19];
  const float* mfc1w = (const float*)d_in[20];
  const float* mfc1b = (const float*)d_in[21];
  const float* mfc2w = (const float*)d_in[22];
  const float* mfc2b = (const float*)d_in[23];

  char* wp = (char*)d_ws;
  auto take = [&](size_t bytes) { void* r = (void*)wp; wp += (bytes + 255) & ~(size_t)255; return r; };

  unsigned short* xb      = (unsigned short*)take((size_t)S_TOK * PVD * 2);
  unsigned short* pe_wb   = (unsigned short*)take((size_t)PVD * HID * 2);
  unsigned short* qkv_wb  = (unsigned short*)take((size_t)NLAYER * HID * 3 * HID * 2);
  unsigned short* proj_wb = (unsigned short*)take((size_t)NLAYER * HID * HID * 2);
  unsigned short* fc1_wb  = (unsigned short*)take((size_t)NLAYER * HID * FFD * 2);
  unsigned short* fc2_wb  = (unsigned short*)take((size_t)NLAYER * FFD * HID * 2);
  unsigned short* mfc1wb  = (unsigned short*)take((size_t)MHID * MHID * 2);
  unsigned short* mfc2wb  = (unsigned short*)take((size_t)MHID * OUTD * 2);
  float*          h       = (float*)take((size_t)S_TOK * HID * 4);
  unsigned short* xnb     = (unsigned short*)take((size_t)S_TOK * HID * 2);
  unsigned short* qkvb    = (unsigned short*)take((size_t)S_TOK * 3 * HID * 2);
  unsigned short* attnb   = (unsigned short*)take((size_t)S_TOK * HID * 2);
  unsigned short* ffab    = (unsigned short*)take((size_t)S_TOK * FFD * 2);

  auto cvt = [&](const float* src, unsigned short* dst, size_t n) {
    cvt_kernel<<<dim3((unsigned)((n + 255) / 256)), dim3(256), 0, stream>>>(src, dst, n);
  };
  cvt(x,     xb,      (size_t)S_TOK * PVD);
  cvt(pe_w,  pe_wb,   (size_t)PVD * HID);
  cvt(qkv_w, qkv_wb,  (size_t)NLAYER * HID * 3 * HID);
  cvt(proj_w,proj_wb, (size_t)NLAYER * HID * HID);
  cvt(fc1_w, fc1_wb,  (size_t)NLAYER * HID * FFD);
  cvt(fc2_w, fc2_wb,  (size_t)NLAYER * FFD * HID);
  cvt(mfc1w, mfc1wb,  (size_t)MHID * MHID);
  cvt(mfc2w, mfc2wb,  (size_t)MHID * OUTD);

  // patch embed: h = x @ pe_w + pe_b
  gemm_bf16_kernel<0><<<dim3(HID / 128, S_TOK / 128), 256, 0, stream>>>(
      xb, pe_wb, pe_b, nullptr, h, nullptr, S_TOK, HID, PVD);

  for (int l = 0; l < NLAYER; l++) {
    ln_bf16_kernel<<<S_TOK, 256, 0, stream>>>(h, ln1_w + (size_t)l * HID, ln1_b + (size_t)l * HID, xnb, HID);
    gemm_bf16_kernel<2><<<dim3(3 * HID / 128, S_TOK / 128), 256, 0, stream>>>(
        xnb, qkv_wb + (size_t)l * HID * 3 * HID, qkv_b + (size_t)l * 3 * HID,
        nullptr, nullptr, qkvb, S_TOK, 3 * HID, HID);
    rope_kernel<<<(S_TOK * 2 * NHEAD * (HD / 2)) / 256, 256, 0, stream>>>(qkvb, cosT, sinT);
    attn_kernel<<<(S_TOK / WINSZ) * NHEAD, 128, 0, stream>>>(qkvb, attnb);
    gemm_bf16_kernel<1><<<dim3(HID / 128, S_TOK / 128), 256, 0, stream>>>(
        attnb, proj_wb + (size_t)l * HID * HID, proj_b + (size_t)l * HID,
        h, h, nullptr, S_TOK, HID, HID);
    ln_bf16_kernel<<<S_TOK, 256, 0, stream>>>(h, ln2_w + (size_t)l * HID, ln2_b + (size_t)l * HID, xnb, HID);
    gemm_bf16_kernel<3><<<dim3(FFD / 128, S_TOK / 128), 256, 0, stream>>>(
        xnb, fc1_wb + (size_t)l * HID * FFD, fc1_b + (size_t)l * FFD,
        nullptr, nullptr, ffab, S_TOK, FFD, HID);
    gemm_bf16_kernel<1><<<dim3(HID / 128, S_TOK / 128), 256, 0, stream>>>(
        ffab, fc2_wb + (size_t)l * FFD * HID, fc2_b + (size_t)l * HID,
        h, h, nullptr, S_TOK, HID, FFD);
  }

  // merger: LN -> (4096,5120) @ m_fc1 -> gelu -> @ m_fc2 -> out f32
  ln_bf16_kernel<<<S_TOK, 256, 0, stream>>>(h, mlnw, mlnb, xnb, HID);
  gemm_bf16_kernel<3><<<dim3(MHID / 128, MROWS / 128), 256, 0, stream>>>(
      xnb, mfc1wb, mfc1b, nullptr, nullptr, ffab, MROWS, MHID, MHID);
  gemm_bf16_kernel<0><<<dim3(OUTD / 128, MROWS / 128), 256, 0, stream>>>(
      ffab, mfc2wb, mfc2b, nullptr, (float*)d_out, nullptr, MROWS, OUTD, MHID);
}
